// GSC_DH_SNN_86474871538225
// MI455X (gfx1250) — matrix operand order repaired and checked by reading, not compiled, not run on hardware
//
#include <hip/hip_runtime.h>
#include <hip/hip_bf16.h>
#include <stdint.h>
#include <stddef.h>

// Problem dims (match reference)
#define B_SZ    64
#define T_STEPS 150
#define IN_DIM  120
#define IN_PAD  128
#define HID     1024
#define OUT_DIM 35
#define NB      8
#define NTOT    (HID * NB)   // 8192
#define VTH     1.0f

// GEMM tiling: each wave owns 16 output columns x ALL 64 batch rows
// (4 accumulators). Workgroup = 4 waves = 64 columns. Grid = NTOT/64 = 128.
#define WG_COLS   64
#define GEMM_BLKS (NTOT / WG_COLS)   // 128
#define RO_BLKS   ((B_SZ * OUT_DIM + 127) / 128)  // 18

typedef __attribute__((ext_vector_type(16))) _Float16 v16h;
typedef __attribute__((ext_vector_type(8)))  _Float16 v8h;
typedef __attribute__((ext_vector_type(8)))  float    v8f;

__device__ __forceinline__ float sigmoid_(float x) {
    return 1.0f / (1.0f + __expf(-x));
}

// ---------------------------------------------------------------------------
// Prologue: zero the state region (d, mem, spk, mr, acc)
// ---------------------------------------------------------------------------
__global__ void zero_ws_kernel(uint32_t* __restrict__ p, size_t n32) {
    size_t i = (size_t)blockIdx.x * blockDim.x + threadIdx.x;
    if (i < n32) p[i] = 0u;
}

// ---------------------------------------------------------------------------
// Prologue: pack masked weights f32 -> f16, pad K to KP with zeros.
// w: [N,K] f32, mask: [N,K] u8 (numpy bool), dst: [N,KP] f16
// ---------------------------------------------------------------------------
__global__ void pack_w_kernel(const float* __restrict__ w,
                              const unsigned char* __restrict__ mask,
                              _Float16* __restrict__ dst,
                              int K, int KP, int N) {
    int i = blockIdx.x * blockDim.x + threadIdx.x;
    if (i >= N * KP) return;
    int n = i / KP, k = i % KP;
    float v = 0.0f;
    if (k < K) {
        float m = mask[(size_t)n * K + k] ? 1.0f : 0.0f;
        v = w[(size_t)n * K + k] * m;
    }
    dst[i] = (_Float16)v;
}

// ---------------------------------------------------------------------------
// Prologue: x [B,1,T,IN] f32 -> xt [T][B][IN_PAD] f16 (zero padded)
// ---------------------------------------------------------------------------
__global__ void pack_x_kernel(const float* __restrict__ x, _Float16* __restrict__ xt) {
    int i = blockIdx.x * blockDim.x + threadIdx.x;
    const int TOT = T_STEPS * B_SZ * IN_PAD;
    if (i >= TOT) return;
    int t = i / (B_SZ * IN_PAD);
    int r = i % (B_SZ * IN_PAD);
    int b = r / IN_PAD;
    int k = r % IN_PAD;
    float v = (k < IN_DIM) ? x[((size_t)b * T_STEPS + t) * IN_DIM + k] : 0.0f;
    xt[i] = (_Float16)v;
}

// ---------------------------------------------------------------------------
// Fused GEMM epilogue for one 16x16 WMMA tile at (rows b0..b0+15, cols n_t..+15):
//   branch decay d' = beta*d + (1-beta)*dense, sum over NB=8 branches,
//   membrane update with soft reset, spike threshold.
// C layout (f32 16x16): lane l holds col = l&15, VGPR v holds row = v + 8*(l>>4).
// ---------------------------------------------------------------------------
__device__ __forceinline__ void tile_epilogue(
    v8f c, int n_t, int lane, int b0,
    const float* __restrict__ tau_n, const float* __restrict__ tau_m,
    float* __restrict__ dstate, float* __restrict__ mem,
    _Float16* __restrict__ spk)
{
    const int l16   = lane & 15;
    const int rbase = (lane >> 4) * 8;      // 0 or 8
    const int n     = n_t + l16;            // global branch column
    const float beta = sigmoid_(tau_n[n]);  // tau_n is [HID*NB] flat, index == n

    float s[8];
    #pragma unroll
    for (int v = 0; v < 8; ++v) {
        int b = b0 + rbase + v;
        float* dp = dstate + (size_t)b * NTOT + n;
        float dn = beta * (*dp) + (1.0f - beta) * c[v];
        *dp = dn;
        s[v] = dn;
    }
    // Sum over the 8 branch columns of each neuron: lane groups {0-7},{8-15},{16-23},{24-31}
    #pragma unroll
    for (int v = 0; v < 8; ++v) {
        s[v] += __shfl_xor(s[v], 1, 32);
        s[v] += __shfl_xor(s[v], 2, 32);
        s[v] += __shfl_xor(s[v], 4, 32);
    }
    // Each lane performs one (batch-row, neuron) membrane update.
    int vsel = lane & 7;
    float li = s[0];
    li = (vsel == 1) ? s[1] : li;
    li = (vsel == 2) ? s[2] : li;
    li = (vsel == 3) ? s[3] : li;
    li = (vsel == 4) ? s[4] : li;
    li = (vsel == 5) ? s[5] : li;
    li = (vsel == 6) ? s[6] : li;
    li = (vsel == 7) ? s[7] : li;

    int h   = (n_t >> 3) + ((l16 >> 3) & 1);   // neuron index
    int row = (lane & 7) + rbase;              // batch row within tile
    int bi  = b0 + row;
    float alpha = sigmoid_(tau_m[h]);
    size_t mi = (size_t)bi * HID + h;
    float m_old = mem[mi];
    float s_old = (float)spk[mi];              // previous-timestep spike
    float m_new = m_old * alpha + (1.0f - alpha) * li - VTH * s_old;
    mem[mi] = m_new;
    spk[mi] = (_Float16)((m_new - VTH > 0.0f) ? 1.0f : 0.0f);
}

// ---------------------------------------------------------------------------
// One recurrent layer for one timestep, optionally fused with the (independent)
// readout update for the PREVIOUS timestep in trailing blocks.
//
//   A  : [B_SZ, lda] f16 activations (spikes or packed input), lda = Kp
//   Wp : [NTOT, ldb] f16 pre-masked weights, ldb = Kp
//   GEMM blocks [0, gemm_blocks): 4 waves x 16 columns, each wave covers ALL
//     64 batch rows (4 C tiles) -> every weight element read from L2 once.
//   Blocks [gemm_blocks, ...): readout mr/acc update if ro_enable.
// ---------------------------------------------------------------------------
__global__ void __launch_bounds__(128) layer_step_kernel(
    const _Float16* __restrict__ A, int lda,
    const _Float16* __restrict__ Wp, int ldb, int Kp,
    const float* __restrict__ bias,
    const float* __restrict__ tau_n,
    const float* __restrict__ tau_m,
    float* __restrict__ dstate,
    float* __restrict__ mem,
    _Float16* __restrict__ spk,
    int gemm_blocks, int ro_enable,
    const _Float16* __restrict__ ro_spk,
    const float* __restrict__ wr,
    const float* __restrict__ br,
    const float* __restrict__ tau_mr,
    float* __restrict__ mr,
    float* __restrict__ accR)
{
    if ((int)blockIdx.x >= gemm_blocks) {
        // ---- fused readout for previous timestep ----
        if (!ro_enable) return;
        int i = ((int)blockIdx.x - gemm_blocks) * (int)blockDim.x + (int)threadIdx.x;
        if (i >= B_SZ * OUT_DIM) return;
        int b = i / OUT_DIM, o = i % OUT_DIM;
        const _Float16* sp = ro_spk + (size_t)b * HID;
        const float*    wp = wr     + (size_t)o * HID;
        float dot = 0.0f;
        #pragma unroll 4
        for (int k = 0; k < HID; ++k) dot += (float)sp[k] * wp[k];
        dot += br[o];
        float ar = sigmoid_(tau_mr[o]);
        float m  = mr[i] * ar + (1.0f - ar) * dot;
        mr[i]   = m;
        accR[i] += m;
        return;
    }

    // ---- GEMM + fused SNN state update ----
    const int lane = threadIdx.x & 31;
    const int wave = threadIdx.x >> 5;
    const int n0   = (int)blockIdx.x * WG_COLS + wave * 16;  // 16 columns per wave

    const int l16  = lane & 15;
    const int aoff = (lane < 16) ? 0 : 16;   // A K-offset (contiguous, matches B)
    const int boff = (lane < 16) ? 0 : 16;   // B K-offset (contiguous, matches A)

    // C init = per-column bias broadcast (folds "+ b" into the GEMM)
    float bv = bias[n0 + l16];
    v8f c0, c1, c2, c3;
    #pragma unroll
    for (int v = 0; v < 8; ++v) { c0[v] = bv; c1[v] = bv; c2[v] = bv; c3[v] = bv; }

    // A row pointers for the 4 M-tiles (rows 0-15, 16-31, 32-47, 48-63)
    const _Float16* aR0 = A + (size_t)( 0 + l16) * lda + aoff;
    const _Float16* aR1 = A + (size_t)(16 + l16) * lda + aoff;
    const _Float16* aR2 = A + (size_t)(32 + l16) * lda + aoff;
    const _Float16* aR3 = A + (size_t)(48 + l16) * lda + aoff;
    const _Float16* wRow = Wp + (size_t)(n0 + l16) * ldb + boff;

    for (int k0 = 0; k0 < Kp; k0 += 32) {
        // B fragment: lane holds column n0+l16, 16 contiguous K values
        v16h bf = *(const v16h*)(wRow + k0);

        // A fragments: lane holds row m=l16 of each M-tile,
        // halves K = k0+aoff+0..7 and k0+aoff+8..15
        v8h a0lo = *(const v8h*)(aR0 + k0);
        v8h a0hi = *(const v8h*)(aR0 + k0 + 8);
        v8h a1lo = *(const v8h*)(aR1 + k0);
        v8h a1hi = *(const v8h*)(aR1 + k0 + 8);
        v8h a2lo = *(const v8h*)(aR2 + k0);
        v8h a2hi = *(const v8h*)(aR2 + k0 + 8);
        v8h a3lo = *(const v8h*)(aR3 + k0);
        v8h a3hi = *(const v8h*)(aR3 + k0 + 8);
        v16h a0, a1, a2, a3;
        #pragma unroll
        for (int i = 0; i < 8; ++i) {
            a0[i] = a0lo[i]; a0[8 + i] = a0hi[i];
            a1[i] = a1lo[i]; a1[8 + i] = a1hi[i];
            a2[i] = a2lo[i]; a2[8 + i] = a2hi[i];
            a3[i] = a3lo[i]; a3[8 + i] = a3hi[i];
        }

        c0 = __builtin_amdgcn_wmma_f32_16x16x32_f16(false, a0, false, bf, (short)0, c0, false, false);
        c1 = __builtin_amdgcn_wmma_f32_16x16x32_f16(false, a1, false, bf, (short)0, c1, false, false);
        c2 = __builtin_amdgcn_wmma_f32_16x16x32_f16(false, a2, false, bf, (short)0, c2, false, false);
        c3 = __builtin_amdgcn_wmma_f32_16x16x32_f16(false, a3, false, bf, (short)0, c3, false, false);
    }

    tile_epilogue(c0, n0, lane,  0, tau_n, tau_m, dstate, mem, spk);
    tile_epilogue(c1, n0, lane, 16, tau_n, tau_m, dstate, mem, spk);
    tile_epilogue(c2, n0, lane, 32, tau_n, tau_m, dstate, mem, spk);
    tile_epilogue(c3, n0, lane, 48, tau_n, tau_m, dstate, mem, spk);
}

// ---------------------------------------------------------------------------
// Final: out = log_softmax(acc / T) per batch row.
// ---------------------------------------------------------------------------
__global__ void finalize_kernel(const float* __restrict__ accR, float* __restrict__ out) {
    int b = threadIdx.x;
    if (b >= B_SZ) return;
    const float* a = accR + (size_t)b * OUT_DIM;
    const float invT = 1.0f / (float)T_STEPS;
    float mx = -3.0e38f;
    for (int o = 0; o < OUT_DIM; ++o) mx = fmaxf(mx, a[o] * invT);
    float se = 0.0f;
    for (int o = 0; o < OUT_DIM; ++o) se += __expf(a[o] * invT - mx);
    float lse = mx + __logf(se);
    for (int o = 0; o < OUT_DIM; ++o) out[(size_t)b * OUT_DIM + o] = a[o] * invT - lse;
}

// ---------------------------------------------------------------------------
// Host launcher
// ---------------------------------------------------------------------------
extern "C" void kernel_launch(void* const* d_in, const int* in_sizes, int n_in,
                              void* d_out, int out_size, void* d_ws, size_t ws_size,
                              hipStream_t stream)
{
    (void)in_sizes; (void)n_in; (void)out_size; (void)ws_size;

    const float* x      = (const float*)d_in[0];
    const float* w1     = (const float*)d_in[1];
    const float* b1     = (const float*)d_in[2];
    const float* tau_m1 = (const float*)d_in[3];
    const float* tau_n1 = (const float*)d_in[4];
    const float* w2     = (const float*)d_in[5];
    const float* b2     = (const float*)d_in[6];
    const float* tau_m2 = (const float*)d_in[7];
    const float* tau_n2 = (const float*)d_in[8];
    const float* w3     = (const float*)d_in[9];
    const float* b3     = (const float*)d_in[10];
    const float* tau_m3 = (const float*)d_in[11];
    const float* tau_n3 = (const float*)d_in[12];
    const float* wr     = (const float*)d_in[13];
    const float* br     = (const float*)d_in[14];
    const float* tau_mr = (const float*)d_in[15];
    const unsigned char* mask1 = (const unsigned char*)d_in[16];
    const unsigned char* mask2 = (const unsigned char*)d_in[17];
    const unsigned char* mask3 = (const unsigned char*)d_in[18];

    // Workspace layout (all chunks 256B-aligned by construction)
    char* ws = (char*)d_ws;
    size_t off = 0;
    auto take = [&](size_t bytes) { size_t o = off; off += (bytes + 255) & ~(size_t)255; return o; };

    _Float16* w1p = (_Float16*)(ws + take((size_t)NTOT * IN_PAD * 2));   // 2 MB
    _Float16* w2p = (_Float16*)(ws + take((size_t)NTOT * HID * 2));      // 16 MB
    _Float16* w3p = (_Float16*)(ws + take((size_t)NTOT * HID * 2));      // 16 MB
    _Float16* xt  = (_Float16*)(ws + take((size_t)T_STEPS * B_SZ * IN_PAD * 2)); // 2.4 MB

    size_t state_begin = off;
    float* d1   = (float*)(ws + take((size_t)B_SZ * NTOT * 4));
    float* d2   = (float*)(ws + take((size_t)B_SZ * NTOT * 4));
    float* d3   = (float*)(ws + take((size_t)B_SZ * NTOT * 4));
    float* mem1 = (float*)(ws + take((size_t)B_SZ * HID * 4));
    float* mem2 = (float*)(ws + take((size_t)B_SZ * HID * 4));
    float* mem3 = (float*)(ws + take((size_t)B_SZ * HID * 4));
    float* mr   = (float*)(ws + take((size_t)B_SZ * OUT_DIM * 4));
    float* accR = (float*)(ws + take((size_t)B_SZ * OUT_DIM * 4));
    _Float16* spk1 = (_Float16*)(ws + take((size_t)B_SZ * HID * 2));
    _Float16* spk2 = (_Float16*)(ws + take((size_t)B_SZ * HID * 2));
    _Float16* spk3 = (_Float16*)(ws + take((size_t)B_SZ * HID * 2));
    size_t state_end = off;

    // 1) Zero all recurrent state (deterministic re-init every call)
    {
        size_t n32 = (state_end - state_begin) / 4;
        int blocks = (int)((n32 + 255) / 256);
        zero_ws_kernel<<<blocks, 256, 0, stream>>>((uint32_t*)(ws + state_begin), n32);
    }

    // 2) Pack masked weights to f16 and the input sequence to padded f16
    {
        int n1 = NTOT * IN_PAD;
        pack_w_kernel<<<(n1 + 255) / 256, 256, 0, stream>>>(w1, mask1, w1p, IN_DIM, IN_PAD, NTOT);
        int n2 = NTOT * HID;
        pack_w_kernel<<<(n2 + 255) / 256, 256, 0, stream>>>(w2, mask2, w2p, HID, HID, NTOT);
        pack_w_kernel<<<(n2 + 255) / 256, 256, 0, stream>>>(w3, mask3, w3p, HID, HID, NTOT);
        int nx = T_STEPS * B_SZ * IN_PAD;
        pack_x_kernel<<<(nx + 255) / 256, 256, 0, stream>>>(x, xt);
    }

    // 3) Recurrence: 150 timesteps x 3 WMMA layer kernels.
    //    The readout for step t-1 rides along in layer1's trailing blocks at step t.
    for (int t = 0; t < T_STEPS; ++t) {
        const _Float16* xin = xt + (size_t)t * B_SZ * IN_PAD;
        layer_step_kernel<<<GEMM_BLKS + RO_BLKS, 128, 0, stream>>>(
            xin, IN_PAD, w1p, IN_PAD, IN_PAD, b1, tau_n1, tau_m1, d1, mem1, spk1,
            GEMM_BLKS, (t > 0) ? 1 : 0, spk3, wr, br, tau_mr, mr, accR);
        layer_step_kernel<<<GEMM_BLKS, 128, 0, stream>>>(
            spk1, HID, w2p, HID, HID, b2, tau_n2, tau_m2, d2, mem2, spk2,
            GEMM_BLKS, 0, (const _Float16*)nullptr, nullptr, nullptr, nullptr, nullptr, nullptr);
        layer_step_kernel<<<GEMM_BLKS, 128, 0, stream>>>(
            spk2, HID, w3p, HID, HID, b3, tau_n3, tau_m3, d3, mem3, spk3,
            GEMM_BLKS, 0, (const _Float16*)nullptr, nullptr, nullptr, nullptr, nullptr, nullptr);
    }
    // Trailing readout for t = T-1 (reuse kernel with gemm_blocks = 0)
    layer_step_kernel<<<RO_BLKS, 128, 0, stream>>>(
        xt, IN_PAD, w1p, IN_PAD, IN_PAD, b1, tau_n1, tau_m1, d1, mem1, spk1,
        0, 1, spk3, wr, br, tau_mr, mr, accR);

    // 4) log_softmax(acc / T)
    finalize_kernel<<<1, 64, 0, stream>>>(accR, (float*)d_out);
}
